// GAT_75265006895713
// MI455X (gfx1250) — compile-verified, matrix operand order checked by reading, not run on hardware
//
#include <hip/hip_runtime.h>
#include <hip/hip_bf16.h>
#include <hip/hip_fp16.h>

// ---------------- problem constants (from reference) ----------------
constexpr int NN   = 100000;   // nodes
constexpr int EE   = 800000;   // edges (before self loops)
constexpr int HH   = 4;        // heads
constexpr int CC   = 64;       // channels per head
constexpr int GG   = 2000;     // graphs
constexpr int FOUT = HH * CC;  // 256
constexpr int EP   = EE + NN;  // edges + self loops = 900000
constexpr float NEG_SLOPE = 0.2f;

typedef __attribute__((ext_vector_type(16))) _Float16 v16h;
typedef __attribute__((ext_vector_type(8)))  float    v8f;
typedef __attribute__((ext_vector_type(8)))  _Float16 h8;
typedef __attribute__((ext_vector_type(4)))  _Float16 h4;

union V16 { v16h v; h8 h[2]; };

// ---------------- monotonic float<->uint encoding for atomicMax ----------------
__device__ __forceinline__ unsigned fenc(float f) {
    unsigned u = __float_as_uint(f);
    return (u & 0x80000000u) ? ~u : (u | 0x80000000u);
}
__device__ __forceinline__ float fdec(unsigned u) {
    u = (u & 0x80000000u) ? (u & 0x7FFFFFFFu) : ~u;
    return __uint_as_float(u);
}

// ---------------- P = W * blockdiag(a_src, a_dst): [FIN, 8] ----------------
// P[k][2*hd+0] = sum_c W[k, hd*64+c] * a_s[hd*64+c]   (same for a_d at +1)
__global__ void gat_compute_P(const float* __restrict__ W,
                              const float* __restrict__ a_s,
                              const float* __restrict__ a_d,
                              float* __restrict__ P, int Fin)
{
    const int t = blockIdx.x * blockDim.x + threadIdx.x;
    if (t >= Fin * 8) return;
    const int k  = t >> 3;
    const int j  = t & 7;
    const int hd = j >> 1;
    const float* a = (j & 1) ? a_d : a_s;
    float s = 0.0f;
#pragma unroll 8
    for (int c = 0; c < CC; ++c)
        s += W[(size_t)k * FOUT + hd * CC + c] * a[hd * CC + c];
    P[k * 8 + j] = s;
}

// ---------------- logits: [als|ald] = X @ P  ([N,FIN]x[FIN,8]) ----------------
__global__ void gat_logits(const float* __restrict__ X,
                           const float* __restrict__ P,
                           float* __restrict__ als,    // [N,H]
                           float* __restrict__ ald,    // [N,H]
                           int Fin)
{
    const int t = blockIdx.x * blockDim.x + threadIdx.x;
    if (t >= NN * 8) return;
    const int n  = t >> 3;
    const int j  = t & 7;
    const int hd = j >> 1;
    float s = 0.0f;
    for (int k = 0; k < Fin; ++k)
        s += X[(size_t)n * Fin + k] * P[k * 8 + j];
    if (j & 1) ald[n * HH + hd] = s;
    else       als[n * HH + hd] = s;
}

// ---------------- WMMA node transform ----------------
// Block = 128 threads (4 waves) = one 16-row x 64-col output tile.
// Tiles of X and W staged into LDS as f16 (W transposed); each wave builds its
// fragments with four 16B ds_load_b128 per k-step, then v_wmma_f32_16x16x32_f16.
constexpr int ASTR = 40;   // sA row stride (halves): 80B, 16B aligned, bank-spread
constexpr int BSTR = 40;   // sB col stride (halves)

template<int FIN>
__global__ __launch_bounds__(128)
void gat_transform_wmma(const float* __restrict__ X,
                        const float* __restrict__ W,     // [FIN, 256]
                        _Float16*    __restrict__ Hout)  // [N, 256] f16
{
    constexpr int KSTEPS = (FIN + 31) / 32;
    __shared__ __align__(16) _Float16 sA[16 * ASTR];   // [row][k]
    __shared__ __align__(16) _Float16 sB[64 * BSTR];   // [col][k] (transposed)

    const int tid    = threadIdx.x;
    const int lane   = tid & 31;
    const int wave   = tid >> 5;
    const int mBase  = blockIdx.x * 16;                // N is a multiple of 16
    const int nBase0 = blockIdx.y * 64;                // block covers 64 columns
    const int col    = lane & 15;
    const int khalf  = lane >> 4;                      // 0 or 1

    v8f c = {};
#pragma unroll
    for (int ks = 0; ks < KSTEPS; ++ks) {
        const int k0 = ks * 32;
        if (ks) __syncthreads();                       // protect LDS reuse

        // ---- stage A: 16 rows x 32 k   (128 thr x 4 elems, f32 -> f16) ----
        {
            const int row = tid >> 3;                  // 0..15
            const int kq  = (tid & 7) * 4;             // 0,4,...,28
            const float* src = X + (size_t)(mBase + row) * FIN + k0 + kq;
            float4 v;
            if constexpr (FIN % 32 == 0) {
                v = *(const float4*)src;               // unconditional, 16B
            } else {
                v.x = (k0 + kq + 0 < FIN) ? src[0] : 0.0f;
                v.y = (k0 + kq + 1 < FIN) ? src[1] : 0.0f;
                v.z = (k0 + kq + 2 < FIN) ? src[2] : 0.0f;
                v.w = (k0 + kq + 3 < FIN) ? src[3] : 0.0f;
            }
            h4 p = { (_Float16)v.x, (_Float16)v.y, (_Float16)v.z, (_Float16)v.w };
            *(h4*)(sA + row * ASTR + kq) = p;          // one 8B LDS store
        }

        // ---- stage B: 32 k x 64 cols, stored transposed sB[c][k] ----
#pragma unroll
        for (int rep = 0; rep < 4; ++rep) {
            const int eidx = rep * 512 + tid * 4;      // 0..2047
            const int kk   = eidx >> 6;                // 0..31
            const int cc   = eidx & 63;
            float4 v;
            if constexpr (FIN % 32 == 0) {
                v = *(const float4*)(W + (size_t)(k0 + kk) * FOUT + nBase0 + cc);
            } else {
                if (k0 + kk < FIN)
                    v = *(const float4*)(W + (size_t)(k0 + kk) * FOUT + nBase0 + cc);
                else
                    v = make_float4(0.0f, 0.0f, 0.0f, 0.0f);
            }
            sB[(cc + 0) * BSTR + kk] = (_Float16)v.x;
            sB[(cc + 1) * BSTR + kk] = (_Float16)v.y;
            sB[(cc + 2) * BSTR + kk] = (_Float16)v.z;
            sB[(cc + 3) * BSTR + kk] = (_Float16)v.w;
        }
        __syncthreads();

        // ---- build fragments (ISA 16-bit A layout) + WMMA ----
        // element i of fragment -> K = 16*(i/8) + 8*khalf + (i%8)
        const int myCol = wave * 16 + col;
        V16 a, b;
        a.h[0] = *(const h8*)(sA + col * ASTR + (khalf << 3));
        a.h[1] = *(const h8*)(sA + col * ASTR + 16 + (khalf << 3));
        b.h[0] = *(const h8*)(sB + myCol * BSTR + (khalf << 3));
        b.h[1] = *(const h8*)(sB + myCol * BSTR + 16 + (khalf << 3));
        c = __builtin_amdgcn_wmma_f32_16x16x32_f16(false, a.v, false, b.v,
                                                   (short)0, c, false, false);
    }

    // ---- epilogue: f16 store of the C/D tile ----
    const int nBase = nBase0 + wave * 16;
#pragma unroll
    for (int r = 0; r < 8; ++r) {
        const int orow = mBase + r + (khalf << 3);     // C/D layout row
        Hout[(size_t)orow * FOUT + nBase + col] = (_Float16)c[r];
    }
}

// ---------------- edge kernels ----------------
__device__ __forceinline__ void edge_ends(const int* __restrict__ ei, int e,
                                          int& s, int& d) {
    if (e < EE) { s = ei[e]; d = ei[EE + e]; }
    else        { s = d = e - EE; }                 // self loop
}

__global__ void gat_edge_max(const int* __restrict__ ei,
                             const float* __restrict__ als,
                             const float* __restrict__ ald,
                             unsigned* __restrict__ emax)    // zero-init = -inf
{
    const int t = blockIdx.x * blockDim.x + threadIdx.x;
    if (t >= EP * HH) return;
    const int e = t >> 2, hd = t & 3;
    int s, d; edge_ends(ei, e, s, d);
    float v = als[s * HH + hd] + ald[d * HH + hd];
    v = v > 0.0f ? v : NEG_SLOPE * v;
    atomicMax(&emax[d * HH + hd], fenc(v));
}

__global__ void gat_edge_expsum(const int* __restrict__ ei,
                                const float* __restrict__ als,
                                const float* __restrict__ ald,
                                const unsigned* __restrict__ emax,
                                float* __restrict__ esum,     // zero-init
                                float* __restrict__ ealpha)   // [EP*H]
{
    const int t = blockIdx.x * blockDim.x + threadIdx.x;
    if (t >= EP * HH) return;
    const int e = t >> 2, hd = t & 3;
    int s, d; edge_ends(ei, e, s, d);
    float v = als[s * HH + hd] + ald[d * HH + hd];
    v = v > 0.0f ? v : NEG_SLOPE * v;
    const float mx = fdec(emax[d * HH + hd]);
    const float ex = __expf(v - mx);
    ealpha[t] = ex;
    atomicAdd(&esum[d * HH + hd], ex);
}

// One thread per (edge, head, 8-channel chunk): 128-bit f16 gather +
// 4x GLOBAL_ATOMIC_PK_ADD_F16 (half2) scatter — half the atomic ops/traffic of f32.
__global__ void gat_aggregate(const int* __restrict__ ei,
                              const float* __restrict__ esum,
                              const float* __restrict__ ealpha,
                              const _Float16* __restrict__ Hin,  // [N,256] f16
                              __half2* __restrict__ accum)       // [N,128] half2, zero-init
{
    const int t = blockIdx.x * blockDim.x + threadIdx.x;
    if (t >= EP * HH * 8) return;
    const int chunk = t & 7;
    const int eh    = t >> 3;
    const int e     = eh >> 2, hd = eh & 3;
    int s, d; edge_ends(ei, e, s, d);
    const _Float16* hrow = Hin + (size_t)s * FOUT + hd * CC + (chunk << 3);
    __builtin_prefetch(hrow, 0, 1);                    // global_prefetch_b8: overlap gather
    const float alpha = ealpha[eh] / (esum[d * HH + hd] + 1e-16f);
    __half2* orow = accum + (((size_t)d * FOUT + hd * CC + (chunk << 3)) >> 1);
    h8 hv = *(const h8*)hrow;
#pragma unroll
    for (int c = 0; c < 4; ++c) {
        __half2 pv = __floats2half2_rn((float)hv[2 * c]     * alpha,
                                       (float)hv[2 * c + 1] * alpha);
        atomicAdd(&orow[c], pv);                       // global_atomic_pk_add_f16
    }
}

// head mean + bias + ReLU
__global__ void gat_finalize(const __half* __restrict__ accum,   // [N,256] f16
                             const float* __restrict__ bias,
                             float* __restrict__ act)            // [N,64]
{
    const int t = blockIdx.x * blockDim.x + threadIdx.x;
    if (t >= NN * CC) return;
    const int n = t >> 6, c = t & 63;
    float v = 0.0f;
#pragma unroll
    for (int hd = 0; hd < HH; ++hd)
        v += __half2float(accum[(size_t)n * FOUT + hd * CC + c]);
    v = v * 0.25f + bias[c];
    act[t] = fmaxf(v, 0.0f);
}

// ---------------- pooling + readout ----------------
__global__ void gat_pool(const float* __restrict__ act,
                         const int* __restrict__ batch,
                         float* __restrict__ pooled,   // [G,64] zero-init
                         float* __restrict__ counts)   // [G] zero-init
{
    const int t = blockIdx.x * blockDim.x + threadIdx.x;
    if (t >= NN * CC) return;
    const int n = t >> 6, c = t & 63;
    const int g = batch[n];
    atomicAdd(&pooled[g * CC + c], act[t]);
    if (c == 0) atomicAdd(&counts[g], 1.0f);
}

__global__ void gat_readout(const float* __restrict__ pooled,
                            const float* __restrict__ counts,
                            const float* __restrict__ lw,
                            const float* __restrict__ lb,
                            float* __restrict__ out)   // [G]
{
    const int g = blockIdx.x * blockDim.x + threadIdx.x;
    if (g >= GG) return;
    const float cnt = fmaxf(counts[g], 1.0f);
    float s = 0.0f;
#pragma unroll 8
    for (int c = 0; c < CC; ++c) s += pooled[g * CC + c] * lw[c];
    out[g] = s / cnt + lb[0];
}

// ---------------- host launch ----------------
extern "C" void kernel_launch(void* const* d_in, const int* in_sizes, int n_in,
                              void* d_out, int out_size, void* d_ws, size_t ws_size,
                              hipStream_t stream) {
    const float* x     = (const float*)d_in[0];
    const int*   ei    = (const int*)d_in[1];
    const int*   batch = (const int*)d_in[2];
    const float* Wl[3]  = { (const float*)d_in[3],  (const float*)d_in[7],  (const float*)d_in[11] };
    const float* asl[3] = { (const float*)d_in[4],  (const float*)d_in[8],  (const float*)d_in[12] };
    const float* adl[3] = { (const float*)d_in[5],  (const float*)d_in[9],  (const float*)d_in[13] };
    const float* bl[3]  = { (const float*)d_in[6],  (const float*)d_in[10], (const float*)d_in[14] };
    const float* lw = (const float*)d_in[15];
    const float* lb = (const float*)d_in[16];
    float* out = (float*)d_out;

    // workspace carve-up (256B aligned)
    char*  ws  = (char*)d_ws;
    size_t off = 0;
    auto alloc = [&](size_t bytes) -> void* {
        void* p = ws + off;
        off += (bytes + 255) & ~(size_t)255;
        return p;
    };
    _Float16* h16    = (_Float16*)alloc((size_t)NN * FOUT * sizeof(_Float16));
    __half2*  accum  = (__half2*) alloc((size_t)NN * FOUT * sizeof(__half));
    float*    als    = (float*)   alloc((size_t)NN * HH * sizeof(float));
    float*    ald    = (float*)   alloc((size_t)NN * HH * sizeof(float));
    unsigned* emax   = (unsigned*)alloc((size_t)NN * HH * sizeof(unsigned));
    float*    esum   = (float*)   alloc((size_t)NN * HH * sizeof(float));
    float*    ealpha = (float*)   alloc((size_t)EP * HH * sizeof(float));
    float*    act    = (float*)   alloc((size_t)NN * CC * sizeof(float));
    float*    Pmat   = (float*)   alloc((size_t)64 * 8 * sizeof(float));
    float*    pooled = (float*)   alloc((size_t)GG * CC * sizeof(float));
    float*    counts = (float*)   alloc((size_t)GG * sizeof(float));
    (void)ws_size; (void)in_sizes; (void)n_in; (void)out_size;

    const int tEH  = EP * HH;          // 3.6M  (edge, head)
    const int tAgg = EP * HH * 8;      // 28.8M (edge, head, chunk)
    const int tNC  = NN * CC;          // 6.4M  (node, channel)

    for (int l = 0; l < 3; ++l) {
        const float* X = (l == 0) ? x : act;
        const int Fin  = (l == 0) ? 9 : CC;

        hipMemsetAsync(accum, 0, (size_t)NN * FOUT * sizeof(__half), stream);
        hipMemsetAsync(emax, 0, (size_t)NN * HH * sizeof(unsigned), stream);  // fenc-domain -inf
        hipMemsetAsync(esum, 0, (size_t)NN * HH * sizeof(float), stream);

        // logits via factored P = W * blockdiag(a_s, a_d)
        gat_compute_P<<<2, 256, 0, stream>>>(Wl[l], asl[l], adl[l], Pmat, Fin);
        gat_logits<<<(NN * 8 + 255) / 256, 256, 0, stream>>>(X, Pmat, als, ald, Fin);

        if (l == 0) {
            gat_transform_wmma<9><<<dim3((NN + 15) / 16, 4), 128, 0, stream>>>(
                x, Wl[0], h16);
        } else {
            gat_transform_wmma<64><<<dim3((NN + 15) / 16, 4), 128, 0, stream>>>(
                act, Wl[l], h16);
        }

        gat_edge_max<<<(tEH + 255) / 256, 256, 0, stream>>>(ei, als, ald, emax);
        gat_edge_expsum<<<(tEH + 255) / 256, 256, 0, stream>>>(ei, als, ald, emax, esum, ealpha);
        gat_aggregate<<<(tAgg + 255) / 256, 256, 0, stream>>>(ei, esum, ealpha, h16, accum);
        gat_finalize<<<(tNC + 255) / 256, 256, 0, stream>>>((const __half*)accum, bl[l], act);
    }

    hipMemsetAsync(pooled, 0, (size_t)GG * CC * sizeof(float), stream);
    hipMemsetAsync(counts, 0, (size_t)GG * sizeof(float), stream);
    gat_pool<<<(tNC + 255) / 256, 256, 0, stream>>>(act, batch, pooled, counts);
    gat_readout<<<(GG + 255) / 256, 256, 0, stream>>>(pooled, counts, lw, lb, out);
}
